// GAT_C_89988154786408
// MI455X (gfx1250) — matrix operand order repaired and checked by reading, not compiled, not run on hardware
//
#include <hip/hip_runtime.h>
#include <hip/hip_bf16.h>
#include <math.h>

// ---------------------------------------------------------------------------
// GAT + TopKPooling pipeline for gfx1250 (MI455X, wave32, WMMA).
// Dense GEMMs: f16 operands, f32 accumulate via v_wmma_f32_16x16x32_f16.
// Double-buffered LDS B tile, batched B-fragment loads -> 4 back-to-back
// WMMAs per K-step, one barrier per K-step.
// Sparse segment-softmax aggregation: 3-pass atomic scheme (mem-bound).
// ---------------------------------------------------------------------------

typedef __attribute__((ext_vector_type(16))) _Float16 v16h;
typedef __attribute__((ext_vector_type(8)))  _Float16 v8h;
typedef __attribute__((ext_vector_type(4)))  _Float16 v4h;
typedef __attribute__((ext_vector_type(8)))  float    v8f;
typedef __attribute__((ext_vector_type(4)))  float    v4f;

#define B_GRAPHS 32
#define N_PER    1024
#define E_EDGES  131072
#define HEADS    3
#define CCH      256
#define HC       768
#define K1       820
#define K2       410
#define K3       205
#define NEG_SLOPE 0.2f
#define BPAD     40   // LDS B-tile row pitch in halves (80 B, 16-B aligned)

// ------------------------------- fills / converts ---------------------------
__global__ void __launch_bounds__(256) fill_f32v4_k(v4f* p, float v, long n4) {
  long i = (long)blockIdx.x * blockDim.x + threadIdx.x;
  long s = (long)gridDim.x * blockDim.x;
  v4f vv = {v, v, v, v};
  for (; i < n4; i += s) p[i] = vv;
}
__global__ void __launch_bounds__(256) fill_i32_k(int* p, int v, long n) {
  long i = (long)blockIdx.x * blockDim.x + threadIdx.x;
  long s = (long)gridDim.x * blockDim.x;
  for (; i < n; i += s) p[i] = v;
}
// n4 = element-count / 4 (all our tensors are multiples of 4)
__global__ void __launch_bounds__(256) cvt_f16_k(
    const float* __restrict__ in, _Float16* __restrict__ out, long n4) {
  long i = (long)blockIdx.x * blockDim.x + threadIdx.x;
  long s = (long)gridDim.x * blockDim.x;
  for (; i < n4; i += s) {
    v4f v = *((const v4f*)in + i);
    v4h h;
    h[0] = (_Float16)v[0]; h[1] = (_Float16)v[1];
    h[2] = (_Float16)v[2]; h[3] = (_Float16)v[3];
    *((v4h*)out + i) = h;
  }
}

// ------------------------------- WMMA GEMM ---------------------------------
// C[M,N] = A[M,K] @ B[K,N] (+ bias[N]).  A,B are f16 row-major, C f32.
// K%32==0, N%64==0, M%16==0.  Block = 256 threads = 8 waves;
// block tile 128(M) x 64(N); wave tile 16x64; B staged transposed in LDS,
// double-buffered; one workgroup barrier per K-step.
__global__ void __launch_bounds__(256) gemm_wmma_f16(
    const _Float16* __restrict__ A, const _Float16* __restrict__ Bm,
    const float* __restrict__ bias, float* __restrict__ C,
    int M, int N, int K) {
  __shared__ _Float16 Bs[2][64][BPAD];       // [buf][n][k], 10 KB
  const int lane = threadIdx.x & 31;
  const int wave = threadIdx.x >> 5;         // 0..7
  const int half = lane >> 4;                // 0 / 1
  const int r    = lane & 15;
  const int tileN  = blockIdx.x * 64;
  const int tileM0 = blockIdx.y * 128 + wave * 16;
  const bool active = tileM0 < M;            // M multiple of 16: exact guard
  const int tileM   = active ? tileM0 : 0;   // keep wave alive for barriers

  const int kk = threadIdx.x >> 3;           // 0..31 : B-tile row (K)
  const int c8 = (threadIdx.x & 7) * 8;      // B-tile col group (N)

  v8f acc[4] = {v8f{}, v8f{}, v8f{}, v8f{}};
  const int nk = K >> 5;

  // prologue: stage tile 0 into buffer 0 (coalesced b128 read, b16 transpose)
  {
    v8h bv = *(const v8h*)(Bm + (size_t)kk * N + tileN + c8);
#pragma unroll
    for (int i = 0; i < 8; ++i) Bs[0][c8 + i][kk] = bv[i];
  }

  for (int kb = 0; kb < nk; ++kb) {
    const int cur = kb & 1;
    const int k0  = kb << 5;
    __syncthreads();                         // Bs[cur] ready; Bs[cur^1] free

    // issue next tile's global read early (overlaps DS reads + WMMAs)
    const bool more = (kb + 1) < nk;
    v8h bnext = v8h{};
    if (more)
      bnext = *(const v8h*)(Bm + (size_t)(k0 + 32 + kk) * N + tileN + c8);

    // A fragment (16x32): lane m=r, kbase=half*8; two b128 loads
    const _Float16* Arow = A + (size_t)(tileM + r) * K + k0 + half * 8;
    __builtin_prefetch(Arow + 32, 0, 1);     // global_prefetch next K block
    v8h alo = *(const v8h*)(Arow);
    v8h ahi = *(const v8h*)(Arow + 16);
    v16h a;
#pragma unroll
    for (int i = 0; i < 8; ++i) { a[i] = alo[i]; a[8 + i] = ahi[i]; }

    // load ALL four B fragments first (8x ds_load_b128), then 4 WMMAs
    v16h bfrag[4];
#pragma unroll
    for (int t = 0; t < 4; ++t) {
      const _Float16* Brow = &Bs[cur][t * 16 + r][half * 8];
      v8h b0 = *(const v8h*)(Brow);
      v8h b1 = *(const v8h*)(Brow + 16);
#pragma unroll
      for (int i = 0; i < 8; ++i) { bfrag[t][i] = b0[i]; bfrag[t][8 + i] = b1[i]; }
    }
#pragma unroll
    for (int t = 0; t < 4; ++t)
      acc[t] = __builtin_amdgcn_wmma_f32_16x16x32_f16(
          false, a, false, bfrag[t], (short)0, acc[t], false, false);

    // write next tile into the other buffer (no barrier needed here)
    if (more) {
#pragma unroll
      for (int i = 0; i < 8; ++i) Bs[cur ^ 1][c8 + i][kk] = bnext[i];
    }
  }

  if (active) {
#pragma unroll
    for (int t = 0; t < 4; ++t) {
      const int n = tileN + t * 16 + r;
      const float bvs = bias ? bias[n] : 0.0f;
#pragma unroll
      for (int i = 0; i < 8; ++i) {
        const int m = tileM + half * 8 + i;  // C/D: row = vgpr + half*8
        C[(size_t)m * N + n] = acc[t][i] + bvs;
      }
    }
  }
}

// ------------------------- attention coefficients ---------------------------
__global__ void __launch_bounds__(256) attn_coeff_k(
    const float* __restrict__ xl, const float* __restrict__ a_s,
    const float* __restrict__ a_d, float* __restrict__ asn,
    float* __restrict__ adn, int n) {
  const int wid  = (int)((blockIdx.x * (long)blockDim.x + threadIdx.x) >> 5);
  const int lane = threadIdx.x & 31;
  const int node = wid / HEADS, h = wid % HEADS;
  if (node >= n) return;
  const float* row = xl + (size_t)node * HC + h * CCH;
  const float* vs  = a_s + h * CCH;
  const float* vd  = a_d + h * CCH;
  float s = 0.f, d = 0.f;
  for (int c = lane; c < CCH; c += 32) {
    const float v = row[c];
    s += v * vs[c];
    d += v * vd[c];
  }
  for (int off = 16; off; off >>= 1) {
    s += __shfl_xor(s, off, 32);
    d += __shfl_xor(d, off, 32);
  }
  if (lane == 0) { asn[node * HEADS + h] = s; adn[node * HEADS + h] = d; }
}

// ----------------------------- edge passes ---------------------------------
__device__ inline void atomicMaxF(float* addr, float val) {
  if (val >= 0.f) atomicMax((int*)addr, __float_as_int(val));
  else            atomicMin((unsigned int*)addr, __float_as_uint(val));
}
__device__ inline float leaky(float v) { return v > 0.f ? v : NEG_SLOPE * v; }

__global__ void __launch_bounds__(256) edge_max_k(
    const int* __restrict__ src, const int* __restrict__ dst,
    const int* __restrict__ mask, const float* __restrict__ asn,
    const float* __restrict__ adn, float* __restrict__ emax, int E, int n) {
  const int e = blockIdx.x * blockDim.x + threadIdx.x;
  if (e >= E + n) return;
  int s, d, m;
  if (e < E) { s = src[e]; d = dst[e]; m = mask ? mask[e] : 1; }
  else       { s = d = e - E; m = 1; }
  if (!m) return;
#pragma unroll
  for (int h = 0; h < HEADS; ++h)
    atomicMaxF(&emax[d * HEADS + h], leaky(asn[s * HEADS + h] + adn[d * HEADS + h]));
}

__global__ void __launch_bounds__(256) edge_exp_k(
    const int* __restrict__ src, const int* __restrict__ dst,
    const int* __restrict__ mask, const float* __restrict__ asn,
    const float* __restrict__ adn, const float* __restrict__ emax,
    float* __restrict__ pbuf, float* __restrict__ den, int E, int n) {
  const int e = blockIdx.x * blockDim.x + threadIdx.x;
  if (e >= E + n) return;
  int s, d, m;
  if (e < E) { s = src[e]; d = dst[e]; m = mask ? mask[e] : 1; }
  else       { s = d = e - E; m = 1; }
#pragma unroll
  for (int h = 0; h < HEADS; ++h) {
    float p = 0.f;
    if (m) {
      const float v = leaky(asn[s * HEADS + h] + adn[d * HEADS + h]);
      p = __expf(v - emax[d * HEADS + h]);
      atomicAdd(&den[d * HEADS + h], p);
    }
    pbuf[(size_t)e * HEADS + h] = p;
  }
}

// one block per edge: agg[d, :] += xl[s, :] * alpha[head]
__global__ void __launch_bounds__(256) edge_agg_k(
    const int* __restrict__ src, const int* __restrict__ dst,
    const int* __restrict__ mask, const float* __restrict__ pbuf,
    const float* __restrict__ den, const float* __restrict__ xl,
    float* __restrict__ agg, int E, int n) {
  const int e = blockIdx.x;
  int s, d, m;
  if (e < E) { s = src[e]; d = dst[e]; m = mask ? mask[e] : 1; }
  else       { s = d = e - E; m = 1; }
  if (!m) return;                                  // uniform per block
  __shared__ float alpha[HEADS];
  if (threadIdx.x < HEADS)
    alpha[threadIdx.x] =
        pbuf[(size_t)e * HEADS + threadIdx.x] / den[d * HEADS + threadIdx.x];
  __syncthreads();
  const float* xr = xl + (size_t)s * HC;
  float* ar = agg + (size_t)d * HC;
  for (int c = threadIdx.x; c < HC; c += 256)
    atomicAdd(&ar[c], xr[c] * alpha[c >> 8]);
}

// bias_eff[j] = bh[j] + sum_k b[k] * Wh[k, j]   (folds (agg+b)@Wh algebra)
__global__ void __launch_bounds__(256) bias_fold_k(
    const float* __restrict__ b, const float* __restrict__ Wh,
    const float* __restrict__ bh, float* __restrict__ beff) {
  const int j = threadIdx.x;
  float s = bh[j];
  for (int k = 0; k < HC; ++k) s += b[k] * Wh[(size_t)k * CCH + j];
  beff[j] = s;
}

// ------------------------------- pooling -----------------------------------
__global__ void __launch_bounds__(256) pool_score_k(
    const float* __restrict__ h, const float* __restrict__ p,
    float* __restrict__ score, int n) {
  const int wid  = (int)((blockIdx.x * (long)blockDim.x + threadIdx.x) >> 5);
  const int lane = threadIdx.x & 31;
  if (wid >= n) return;
  const float* row = h + (size_t)wid * CCH;
  float dp = 0.f, pp = 0.f;
  for (int c = lane; c < CCH; c += 32) {
    const float pv = p[c];
    dp += row[c] * pv;
    pp += pv * pv;
  }
  for (int off = 16; off; off >>= 1) {
    dp += __shfl_xor(dp, off, 32);
    pp += __shfl_xor(pp, off, 32);
  }
  if (lane == 0) score[wid] = tanhf(dp * rsqrtf(pp));
}

// per-graph full bitonic sort (desc by score, asc index tiebreak), emit top-k
__global__ void __launch_bounds__(512) topk_select_k(
    const float* __restrict__ score, int n_per, int k,
    int* __restrict__ gsel, float* __restrict__ gval, int* __restrict__ pos) {
  __shared__ float sv[1024];
  __shared__ int   si[1024];
  const int g = blockIdx.x;
  for (int i = threadIdx.x; i < 1024; i += blockDim.x) {
    if (i < n_per) { sv[i] = score[g * n_per + i]; si[i] = i; }
    else           { sv[i] = -__builtin_inff();    si[i] = 1 << 30; }
  }
  __syncthreads();
  for (int ksz = 2; ksz <= 1024; ksz <<= 1) {
    for (int j = ksz >> 1; j > 0; j >>= 1) {
      for (int i = threadIdx.x; i < 1024; i += blockDim.x) {
        const int ixj = i ^ j;
        if (ixj > i) {
          const bool dir = ((i & ksz) == 0);   // true => descending segment
          const float a = sv[i], b = sv[ixj];
          const int ia = si[i], ib = si[ixj];
          const bool a_first = (a > b) || (a == b && ia < ib);
          if (dir ? !a_first : a_first) {
            sv[i] = b; sv[ixj] = a; si[i] = ib; si[ixj] = ia;
          }
        }
      }
      __syncthreads();
    }
  }
  for (int i = threadIdx.x; i < k; i += blockDim.x) {
    const int node = g * n_per + si[i];
    const int nr   = g * k + i;
    gsel[nr] = node;
    gval[nr] = sv[i];
    pos[node] = nr;
  }
}

__global__ void __launch_bounds__(256) pool_gather_k(
    const float* __restrict__ h, const int* __restrict__ gsel,
    const float* __restrict__ gval, float* __restrict__ xn) {
  const int r = blockIdx.x, c = threadIdx.x;
  xn[(size_t)r * CCH + c] = h[(size_t)gsel[r] * CCH + c] * gval[r];
}

__global__ void __launch_bounds__(256) relabel_k(
    const int* __restrict__ src, const int* __restrict__ dst,
    const int* __restrict__ mask, const int* __restrict__ pos,
    int* __restrict__ osrc, int* __restrict__ odst, int* __restrict__ omask,
    int E) {
  const int e = blockIdx.x * blockDim.x + threadIdx.x;
  if (e >= E) return;
  const int m = mask ? mask[e] : 1;
  const int s = pos[src[e]], d = pos[dst[e]];
  const int ok = (m && s >= 0 && d >= 0) ? 1 : 0;
  osrc[e] = ok ? s : 0;
  odst[e] = ok ? d : 0;
  omask[e] = ok;
}

// -------------------------- readout & head ---------------------------------
__global__ void __launch_bounds__(256) readout_k(
    const float* __restrict__ xn, int k, float* __restrict__ zsum, int accum) {
  const int g = blockIdx.x, c = threadIdx.x;
  float mx = -__builtin_inff(), sm = 0.f;
  for (int i = 0; i < k; ++i) {
    const float v = xn[((size_t)g * k + i) * CCH + c];
    mx = fmaxf(mx, v);
    sm += v;
  }
  const float mean = sm / (float)k;
  if (accum) { zsum[g * 512 + c] += mx; zsum[g * 512 + 256 + c] += mean; }
  else       { zsum[g * 512 + c]  = mx; zsum[g * 512 + 256 + c]  = mean; }
}

__global__ void __launch_bounds__(256) mlp_head_k(
    const float* __restrict__ zsum, const float* __restrict__ Wl1,
    const float* __restrict__ bl1, const float* __restrict__ Wl2,
    const float* __restrict__ bl2, float* __restrict__ out) {
  __shared__ float hred[256];
  const int g = blockIdx.x, j = threadIdx.x;
  const float* z = zsum + g * 512;
  float s = bl1[j];
  for (int i = 0; i < 512; ++i) s += z[i] * Wl1[(size_t)i * 256 + j];
  s = fmaxf(s, 0.f);
  hred[j] = s * Wl2[j];
  __syncthreads();
  for (int off = 128; off; off >>= 1) {
    if (j < off) hred[j] += hred[j + off];
    __syncthreads();
  }
  if (j == 0) out[g] = 1.f / (1.f + __expf(-(hred[0] + bl2[0])));
}

// ---------------------------------------------------------------------------
extern "C" void kernel_launch(void* const* d_in, const int* in_sizes, int n_in,
                              void* d_out, int out_size, void* d_ws,
                              size_t ws_size, hipStream_t stream) {
  (void)in_sizes; (void)n_in; (void)out_size; (void)ws_size;

  const float* X    = (const float*)d_in[0];
  const int*   EIDX = (const int*)  d_in[2];
  const float* W1   = (const float*)d_in[4];
  const float* AS1  = (const float*)d_in[5];
  const float* AD1  = (const float*)d_in[6];
  const float* B1   = (const float*)d_in[7];
  const float* WH1  = (const float*)d_in[8];
  const float* BH1  = (const float*)d_in[9];
  const float* P1   = (const float*)d_in[10];
  const float* W2   = (const float*)d_in[11];
  const float* AS2  = (const float*)d_in[12];
  const float* AD2  = (const float*)d_in[13];
  const float* B2   = (const float*)d_in[14];
  const float* WH2  = (const float*)d_in[15];
  const float* BH2  = (const float*)d_in[16];
  const float* P2   = (const float*)d_in[17];
  const float* W3   = (const float*)d_in[18];
  const float* AS3  = (const float*)d_in[19];
  const float* AD3  = (const float*)d_in[20];
  const float* B3   = (const float*)d_in[21];
  const float* WH3  = (const float*)d_in[22];
  const float* BH3  = (const float*)d_in[23];
  const float* WL1  = (const float*)d_in[24];
  const float* BL1  = (const float*)d_in[25];
  const float* WL2  = (const float*)d_in[26];
  const float* BL2  = (const float*)d_in[27];

  const int n1 = B_GRAPHS * N_PER;   // 32768
  const int n2 = B_GRAPHS * K1;      // 26240
  const int n3 = B_GRAPHS * K2;      // 13120
  const int E  = E_EDGES;
  const int Tmax = E + n1;

  // ---- carve workspace -----------------------------------------------------
  size_t off = 0;
  auto carve = [&](size_t bytes) -> void* {
    void* p = (char*)d_ws + off;
    off += (bytes + 255) & ~(size_t)255;
    return p;
  };
  float* XL   = (float*)carve((size_t)n1 * HC * 4);
  float* AGG  = (float*)carve((size_t)n1 * HC * 4);
  float* Hb   = (float*)carve((size_t)n1 * CCH * 4);
  float* XN   = (float*)carve((size_t)n2 * CCH * 4);
  _Float16* AF16 = (_Float16*)carve((size_t)n1 * HC * 2);
  _Float16* W1H  = (_Float16*)carve((size_t)128 * HC * 2);
  _Float16* W2H  = (_Float16*)carve((size_t)CCH * HC * 2);
  _Float16* W3H  = (_Float16*)carve((size_t)CCH * HC * 2);
  _Float16* WH1H = (_Float16*)carve((size_t)HC * CCH * 2);
  _Float16* WH2H = (_Float16*)carve((size_t)HC * CCH * 2);
  _Float16* WH3H = (_Float16*)carve((size_t)HC * CCH * 2);
  float* ASN  = (float*)carve((size_t)n1 * HEADS * 4);
  float* ADN  = (float*)carve((size_t)n1 * HEADS * 4);
  float* EMAX = (float*)carve((size_t)n1 * HEADS * 4);
  float* DEN  = (float*)carve((size_t)n1 * HEADS * 4);
  float* PBUF = (float*)carve((size_t)Tmax * HEADS * 4);
  float* SCORE= (float*)carve((size_t)n1 * 4);
  int*   GSEL = (int*)  carve((size_t)n2 * 4);
  float* GVAL = (float*)carve((size_t)n2 * 4);
  int*   POS  = (int*)  carve((size_t)n1 * 4);
  int*   SRCW = (int*)  carve((size_t)E * 4);
  int*   DSTW = (int*)  carve((size_t)E * 4);
  int*   MSKW = (int*)  carve((size_t)E * 4);
  float* ZSUM = (float*)carve((size_t)B_GRAPHS * 512 * 4);
  float* BEFF = (float*)carve(256 * 4);

  const float NEG_INF = -__builtin_inff();
  // all float fills here have element counts divisible by 4
  auto fillf = [&](float* p, float v, long n) {
    long n4 = n >> 2;
    int blk = (int)((n4 + 255) / 256); if (blk > 16384) blk = 16384;
    hipLaunchKernelGGL(fill_f32v4_k, dim3(blk), dim3(256), 0, stream,
                       (v4f*)p, v, n4);
  };
  auto filli = [&](int* p, int v, long n) {
    int blk = (int)((n + 255) / 256); if (blk > 16384) blk = 16384;
    hipLaunchKernelGGL(fill_i32_k, dim3(blk), dim3(256), 0, stream, p, v, n);
  };
  auto cvt = [&](const float* in, _Float16* out, long n) {
    long n4 = n >> 2;
    int blk = (int)((n4 + 255) / 256); if (blk > 16384) blk = 16384;
    hipLaunchKernelGGL(cvt_f16_k, dim3(blk), dim3(256), 0, stream, in, out, n4);
  };
  // convert A to f16 scratch, then WMMA GEMM with f16 weights
  auto gemm = [&](const float* A32, const _Float16* B16, const float* bias,
                  float* C, int M, int N, int K) {
    cvt(A32, AF16, (long)M * K);
    dim3 grid(N / 64, (M + 127) / 128);
    hipLaunchKernelGGL(gemm_wmma_f16, grid, dim3(256), 0, stream,
                       AF16, B16, bias, C, M, N, K);
  };

  // weights -> f16 once per launch
  cvt(W1,  W1H,  (long)128 * HC);
  cvt(W2,  W2H,  (long)CCH * HC);
  cvt(W3,  W3H,  (long)CCH * HC);
  cvt(WH1, WH1H, (long)HC * CCH);
  cvt(WH2, WH2H, (long)HC * CCH);
  cvt(WH3, WH3H, (long)HC * CCH);

  // one full GAT block (edge softmax + aggregate + second GEMM)
  auto gat_block = [&](const float* xin, int n, int Kin,
                       const int* src, const int* dst, const int* msk,
                       const _Float16* W16, const float* as_, const float* ad_,
                       const float* b, const float* Wh32, const _Float16* Wh16,
                       const float* bh) {
    gemm(xin, W16, nullptr, XL, n, HC, Kin);
    {
      long waves = (long)n * HEADS;
      int blk = (int)((waves * 32 + 255) / 256);
      hipLaunchKernelGGL(attn_coeff_k, dim3(blk), dim3(256), 0, stream,
                         XL, as_, ad_, ASN, ADN, n);
    }
    fillf(EMAX, NEG_INF, (long)n * HEADS);
    fillf(DEN, 0.f, (long)n * HEADS);
    fillf(AGG, 0.f, (long)n * HC);
    const int T = E + n;
    hipLaunchKernelGGL(edge_max_k, dim3((T + 255) / 256), dim3(256), 0, stream,
                       src, dst, msk, ASN, ADN, EMAX, E, n);
    hipLaunchKernelGGL(edge_exp_k, dim3((T + 255) / 256), dim3(256), 0, stream,
                       src, dst, msk, ASN, ADN, EMAX, PBUF, DEN, E, n);
    hipLaunchKernelGGL(edge_agg_k, dim3(T), dim3(256), 0, stream,
                       src, dst, msk, PBUF, DEN, XL, AGG, E, n);
    hipLaunchKernelGGL(bias_fold_k, dim3(1), dim3(256), 0, stream,
                       b, Wh32, bh, BEFF);
    gemm(AGG, Wh16, BEFF, Hb, n, CCH, HC);
  };

  auto pool_block = [&](int n, int n_per, int k, const float* pvec) {
    int blk = (int)(((long)n * 32 + 255) / 256);
    hipLaunchKernelGGL(pool_score_k, dim3(blk), dim3(256), 0, stream,
                       Hb, pvec, SCORE, n);
    filli(POS, -1, n);
    hipLaunchKernelGGL(topk_select_k, dim3(B_GRAPHS), dim3(512), 0, stream,
                       SCORE, n_per, k, GSEL, GVAL, POS);
    hipLaunchKernelGGL(pool_gather_k, dim3(B_GRAPHS * k), dim3(256), 0, stream,
                       Hb, GSEL, GVAL, XN);
  };

  const int* src0 = EIDX;
  const int* dst0 = EIDX + E;

  // ------------------------------ block 1 -----------------------------------
  gat_block(X, n1, 128, src0, dst0, nullptr,
            W1H, AS1, AD1, B1, WH1, WH1H, BH1);
  pool_block(n1, N_PER, K1, P1);
  hipLaunchKernelGGL(relabel_k, dim3((E + 255) / 256), dim3(256), 0, stream,
                     src0, dst0, (const int*)nullptr, POS, SRCW, DSTW, MSKW, E);
  hipLaunchKernelGGL(readout_k, dim3(B_GRAPHS), dim3(256), 0, stream,
                     XN, K1, ZSUM, 0);

  // ------------------------------ block 2 -----------------------------------
  gat_block(XN, n2, CCH, SRCW, DSTW, MSKW,
            W2H, AS2, AD2, B2, WH2, WH2H, BH2);
  pool_block(n2, K1, K2, P2);
  hipLaunchKernelGGL(relabel_k, dim3((E + 255) / 256), dim3(256), 0, stream,
                     SRCW, DSTW, MSKW, POS, SRCW, DSTW, MSKW, E);
  hipLaunchKernelGGL(readout_k, dim3(B_GRAPHS), dim3(256), 0, stream,
                     XN, K2, ZSUM, 1);

  // ------------------------------ block 3 (pool reuses p2) ------------------
  gat_block(XN, n3, CCH, SRCW, DSTW, MSKW,
            W3H, AS3, AD3, B3, WH3, WH3H, BH3);
  pool_block(n3, K2, K3, P2);
  hipLaunchKernelGGL(readout_k, dim3(B_GRAPHS), dim3(256), 0, stream,
                     XN, K3, ZSUM, 1);

  // ------------------------------ head --------------------------------------
  hipLaunchKernelGGL(mlp_head_k, dim3(B_GRAPHS), dim3(256), 0, stream,
                     ZSUM, WL1, BL1, WL2, BL2, (float*)d_out);
}
